// SelfAttention_47888885350607
// MI455X (gfx1250) — matrix operand order repaired and checked by reading, not compiled, not run on hardware
//
#include <hip/hip_runtime.h>

// ---------------------------------------------------------------------------
// SelfAttention (B=8, C=256, H=W=64 -> N=4096, C8=32) for MI455X / gfx1250.
// ~83 GFLOP vs ~100 MB HBM -> compute bound: bf16 WMMA (f32 accum) flash
// attention; the 512 MB score matrix never exists. K/V tiles staged per-block
// into LDS by the Tensor Data Mover (double buffered, TENSORcnt-tracked),
// using TDM padding (16B per 64B row) to kill LDS bank conflicts; cuts L2
// read traffic 4x vs per-wave global loads.
// ---------------------------------------------------------------------------

typedef __bf16 bf16_t;
typedef __attribute__((ext_vector_type(16))) __bf16 bf16x16;
typedef __attribute__((ext_vector_type(8)))  __bf16 bf16x8;
typedef __attribute__((ext_vector_type(8)))  float  f32x8;
typedef __attribute__((ext_vector_type(4)))  unsigned int u32x4;
typedef __attribute__((ext_vector_type(8)))  int i32x8;
typedef __attribute__((ext_vector_type(4)))  int i32x4;

constexpr int   B_  = 8;
constexpr int   C_  = 256;
constexpr int   N_  = 4096;   // H*W
constexpr int   C8_ = 32;     // q/k dim
constexpr float LOG2E = 1.44269504088896340736f;

// padded LDS row: 32 bf16 data + 8 bf16 pad = 80B -> conflict-free 16-lane reads
constexpr int VROW = 40;

#if defined(__gfx1250__) && __has_builtin(__builtin_amdgcn_tensor_load_to_lds) && \
    __has_builtin(__builtin_amdgcn_s_wait_tensorcnt)
#define USE_TDM 1
#else
#define USE_TDM 0
#endif

// Diagnostic: stderr tells us which data-movement path actually compiled.
#if USE_TDM
#warning "CDNA5 TDM path ENABLED: tensor_load_to_lds + s_wait_tensorcnt"
#else
#warning "CDNA5 TDM path DISABLED: cooperative global->LDS fallback in use"
#endif

// ---------------------------------------------------------------------------
// Kernel 1: QKV projections (1x1 convs == channel matmuls), f32 -> bf16.
//   qT[b][n][d] (d contiguous), kT[b][n][d], v[b][c][n] (n contiguous).
// ---------------------------------------------------------------------------
__global__ __launch_bounds__(128)
void qkv_proj_kernel(const float* __restrict__ x,
                     const float* __restrict__ Wq, const float* __restrict__ bq,
                     const float* __restrict__ Wk, const float* __restrict__ bk,
                     const float* __restrict__ Wv, const float* __restrict__ bv,
                     bf16_t* __restrict__ qT, bf16_t* __restrict__ kT,
                     bf16_t* __restrict__ vbf)
{
  __shared__ bf16_t xT[64][C_];          // [n][c] bf16, 32 KB

  const int b    = blockIdx.y;
  const int n0   = blockIdx.x * 64;
  const int t    = threadIdx.x;
  const int wave = t >> 5;
  const int lane = t & 31;

  for (int rr = 0; rr < 2; ++rr) {
    const int c = 2 * t + rr;
    const float* src = x + ((size_t)b * C_ + c) * N_ + n0;
    #pragma unroll
    for (int nn = 0; nn < 64; nn += 4) {
      const float4 v4 = *(const float4*)(src + nn);
      xT[nn + 0][c] = (bf16_t)v4.x;
      xT[nn + 1][c] = (bf16_t)v4.y;
      xT[nn + 2][c] = (bf16_t)v4.z;
      xT[nn + 3][c] = (bf16_t)v4.w;
    }
  }
  __syncthreads();

  const bool lo   = (lane < 16);
  const int  l16  = lane & 15;
  const int  aoff = lo ? 0 : 8;          // accumulator row offset (rows 8..15 on hi lanes)
  const int  koff = lo ? 0 : 16;         // B-operand K offset
  const int  nloc = wave * 16 + l16;
  const int  nG   = n0 + nloc;

  for (int rt = 0; rt < 20; ++rt) {
    const float* Wrow; const float* brow; int o0; int kind;
    if (rt < 2)      { Wrow = Wq; brow = bq; o0 = rt * 16;       kind = 0; }
    else if (rt < 4) { Wrow = Wk; brow = bk; o0 = (rt - 2) * 16; kind = 1; }
    else             { Wrow = Wv; brow = bv; o0 = (rt - 4) * 16; kind = 2; }

    const int orow = o0 + l16;
    f32x8 acc = {0.f,0.f,0.f,0.f,0.f,0.f,0.f,0.f};

    for (int c0 = 0; c0 < C_; c0 += 32) {
      const float* wp = Wrow + (size_t)orow * C_ + c0;
      const float4 wa = *(const float4*)(wp + koff);
      const float4 wb = *(const float4*)(wp + koff + 4);
      const float4 wc = *(const float4*)(wp + 8 + koff);
      const float4 wd = *(const float4*)(wp + 12 + koff);
      bf16x16 a;
      a[0]  = (bf16_t)wa.x; a[1]  = (bf16_t)wa.y; a[2]  = (bf16_t)wa.z; a[3]  = (bf16_t)wa.w;
      a[4]  = (bf16_t)wb.x; a[5]  = (bf16_t)wb.y; a[6]  = (bf16_t)wb.z; a[7]  = (bf16_t)wb.w;
      a[8]  = (bf16_t)wc.x; a[9]  = (bf16_t)wc.y; a[10] = (bf16_t)wc.z; a[11] = (bf16_t)wc.w;
      a[12] = (bf16_t)wd.x; a[13] = (bf16_t)wd.y; a[14] = (bf16_t)wd.z; a[15] = (bf16_t)wd.w;

      const bf16x16 bx = *(const bf16x16*)&xT[nloc][c0 + koff];
      acc = __builtin_amdgcn_wmma_f32_16x16x32_bf16(false, a, false, bx,
                                                    (short)0, acc, false, false);
    }

    const int obase = o0 + aoff;
    #pragma unroll
    for (int r = 0; r < 8; ++r) acc[r] += brow[obase + r];

    if (kind < 2) {
      bf16_t* dst = (kind == 0 ? qT : kT) + ((size_t)b * N_ + nG) * C8_ + obase;
      bf16x8 pk;
      #pragma unroll
      for (int r = 0; r < 8; ++r) pk[r] = (bf16_t)acc[r];
      *(bf16x8*)dst = pk;
    } else {
      #pragma unroll
      for (int r = 0; r < 8; ++r)
        vbf[((size_t)b * C_ + (obase + r)) * N_ + nG] = (bf16_t)acc[r];
    }
  }
}

#if USE_TDM
// ---------------------------------------------------------------------------
// TDM descriptor build + issue (one wave issues; TENSORcnt tracks).
// D# bitfields per CDNA5 ISA.
// 2D tile: tile_dim0 elems/row (x tile_dim1 rows), row stride in elements.
// pad: insert 4 dwords (16B) after every 16 dwords (64B) stored -> 80B rows.
// ---------------------------------------------------------------------------
__device__ __forceinline__ void tdm_load_tile(unsigned int lds_off, const void* gptr,
                                              unsigned int tile_d0, unsigned int tile_d1,
                                              unsigned int row_stride_elems)
{
  const unsigned long long ga = (unsigned long long)(uintptr_t)gptr;
  u32x4 g0;
  g0[0] = 1u;                                        // count=1 (valid), user mode
  g0[1] = lds_off;                                   // lds_addr (bytes)
  g0[2] = (unsigned int)ga;                          // global_addr[31:0]
  g0[3] = (unsigned int)((ga >> 32) & 0x1FFFFFFu)    // global_addr[56:32]
          | (2u << 30);                              // type=2 ("image")

  i32x8 g1;
  // workgroup_mask=0, data_size=1 (2B), pad_enable, pad_interval=3 (16 dw),
  // pad_amount=3 (4 dw)
  g1[0] = (int)((1u << 16) | (1u << 20) | (3u << 22) | (3u << 25));
  const unsigned int tdim0 = 0x40000000u;            // huge -> no OOB zeroing
  const unsigned int tdim1 = 0x40000000u;
  g1[1] = (int)((tdim0 & 0xFFFFu) << 16);            // [47:32]=atomic_addr=0,[63:48]=tdim0.lo
  g1[2] = (int)((tdim0 >> 16) | ((tdim1 & 0xFFFFu) << 16)); // tdim0.hi | tdim1.lo
  g1[3] = (int)((tdim1 >> 16) | (tile_d0 << 16));    // tdim1.hi | tile_dim0
  g1[4] = (int)(tile_d1 & 0xFFFFu);                  // tile_dim1 | tile_dim2=0
  g1[5] = (int)row_stride_elems;                     // tensor_dim0_stride[31:0]
  g1[6] = 0;                                         // stride hi | dim1_stride lo
  g1[7] = 0;

  const i32x4 z4 = {0, 0, 0, 0};
#if __clang_major__ >= 23
  const i32x8 z8 = {0, 0, 0, 0, 0, 0, 0, 0};
  __builtin_amdgcn_tensor_load_to_lds(g0, g1, z4, z4, z8, 0);
#else
  __builtin_amdgcn_tensor_load_to_lds(g0, g1, z4, z4, 0);
#endif
}
#endif  // USE_TDM

// ---------------------------------------------------------------------------
// Kernel 2: fused flash attention + gamma*out + x.
// Block: 128 threads (4 waves); wave owns 16 queries, all 256 out channels.
// Per 32-j step: 2 WMMAs S^T=K^T*Q, online softmax, 16 WMMAs O^T += V^T*P.
// K/V tiles staged once per block into LDS (TDM double-buffer when available).
// ---------------------------------------------------------------------------
__global__ __launch_bounds__(128, 1)
void attn_kernel(const bf16_t* __restrict__ qT, const bf16_t* __restrict__ kT,
                 const bf16_t* __restrict__ vbf, const float* __restrict__ x,
                 const float* __restrict__ gamma, float* __restrict__ out)
{
#if USE_TDM
  constexpr int NBUF = 2;
#else
  constexpr int NBUF = 1;
#endif
  __shared__ __align__(128) bf16_t Kt[NBUF][32][VROW];   // padded rows (80B)
  __shared__ __align__(128) bf16_t Vt[NBUF][C_][VROW];

  const int  b    = blockIdx.y;
  const int  t    = threadIdx.x;
  const int  wave = t >> 5;
  const int  lane = t & 31;
  const bool lo   = (lane < 16);
  const int  l16  = lane & 15;
  const int  aoff = lo ? 0 : 8;
  const int  icol = blockIdx.x * 64 + wave * 16 + l16;

  const bf16_t* kbase = kT  + (size_t)b * N_ * C8_;
  const bf16_t* vbase = vbf + (size_t)b * C_ * N_;

  // Persistent Q B-operand: col i = icol, K(d) = 0..15 / 16..31 contiguous.
  const bf16x16 Bq =
      *(const bf16x16*)(qT + ((size_t)b * N_ + icol) * C8_ + (lo ? 0 : 16));

  const f32x8 zc = {0.f,0.f,0.f,0.f,0.f,0.f,0.f,0.f};
  float m = -3.0e38f, l = 0.f;           // per-query running max/sum (lane-pair
  f32x8 Ov[16];                          //  replicated); O^T accumulators
  #pragma unroll
  for (int ct = 0; ct < 16; ++ct) Ov[ct] = zc;

#if USE_TDM
  const unsigned int ktOff = (unsigned int)(uintptr_t)&Kt[0][0][0];
  const unsigned int vtOff = (unsigned int)(uintptr_t)&Vt[0][0][0];
  constexpr unsigned int KT_STRIDE = 32 * VROW * 2;    // bytes per K buffer
  constexpr unsigned int VT_STRIDE = C_ * VROW * 2;    // bytes per V buffer

  // prologue: fill buffer 0 for j0 = 0
  if (wave == 0) {
    tdm_load_tile(ktOff, kbase, /*1D*/ 32 * 32, 0, 0);
    tdm_load_tile(vtOff, vbase, 32, C_, N_);
    __builtin_amdgcn_s_wait_tensorcnt(0);
  }
  __syncthreads();
#endif

  for (int j0 = 0; j0 < N_; j0 += 32) {
#if USE_TDM
    const int buf = (j0 >> 5) & 1;
    // prefetch next tile into the other buffer (its readers finished last iter)
    if (wave == 0 && (j0 + 32) < N_) {
      const int nb = buf ^ 1;
      tdm_load_tile(ktOff + nb * KT_STRIDE, kbase + (size_t)(j0 + 32) * C8_,
                    32 * 32, 0, 0);
      tdm_load_tile(vtOff + nb * VT_STRIDE, vbase + (j0 + 32), 32, C_, N_);
    }
#else
    const int buf = 0;
    __syncthreads();   // previous iteration's readers done before overwrite
    {                  // cooperative staging (fallback path)
      const int row = t >> 2, ch = (t & 3) * 8;
      *(bf16x8*)&Kt[0][row][ch] =
          *(const bf16x8*)(kbase + (size_t)(j0 + row) * C8_ + ch);
      #pragma unroll
      for (int i = 0; i < 8; ++i) {
        const int vr = row + i * 32;
        *(bf16x8*)&Vt[0][vr][ch] =
            *(const bf16x8*)(vbase + (size_t)vr * N_ + j0 + ch);
      }
    }
    __syncthreads();
#endif

    // ---- S^T = K^T * Q : two [16j x 16i] tiles (K dim = d = 32 exact) ----
    bf16x16 a0, a1;
    {
      const bf16_t* kr0 = &Kt[buf][l16][0];
      const bf16_t* kr1 = &Kt[buf][16 + l16][0];
      const bf16x8 x0 = *(const bf16x8*)(kr0 + (lo ? 0 : 16));
      const bf16x8 x1 = *(const bf16x8*)(kr0 + 8 + (lo ? 0 : 16));
      const bf16x8 y0 = *(const bf16x8*)(kr1 + (lo ? 0 : 16));
      const bf16x8 y1 = *(const bf16x8*)(kr1 + 8 + (lo ? 0 : 16));
      #pragma unroll
      for (int e = 0; e < 8; ++e) {
        a0[e] = x0[e]; a0[8 + e] = x1[e];
        a1[e] = y0[e]; a1[8 + e] = y1[e];
      }
    }
    f32x8 s0 = __builtin_amdgcn_wmma_f32_16x16x32_bf16(false, a0, false, Bq,
                                                       (short)0, zc, false, false);
    f32x8 s1 = __builtin_amdgcn_wmma_f32_16x16x32_bf16(false, a1, false, Bq,
                                                       (short)0, zc, false, false);

    // ---- online softmax over j: 16 in-lane values + 1 shfl to partner ----
    float tmax = s0[0];
    #pragma unroll
    for (int r = 1; r < 8; ++r) tmax = fmaxf(tmax, s0[r]);
    #pragma unroll
    for (int r = 0; r < 8; ++r) tmax = fmaxf(tmax, s1[r]);
    tmax = fmaxf(tmax, __shfl_xor(tmax, 16));
    const float mnew = fmaxf(m, tmax);
    const float corr = exp2f((m - mnew) * LOG2E);

    float p0[8], p1[8];
    float lsum = 0.f;
    #pragma unroll
    for (int r = 0; r < 8; ++r) {
      p0[r] = exp2f((s0[r] - mnew) * LOG2E);
      p1[r] = exp2f((s1[r] - mnew) * LOG2E);
      lsum += p0[r] + p1[r];
    }
    lsum += __shfl_xor(lsum, 16);
    l = l * corr + lsum;
    m = mnew;

    #pragma unroll
    for (int ct = 0; ct < 16; ++ct) {
      #pragma unroll
      for (int r = 0; r < 8; ++r) Ov[ct][r] *= corr;
    }

    // ---- S^T D-layout -> P B-operand [K=32 j x N=16 i] (1 shfl / element) --
    bf16x16 Bp;
    #pragma unroll
    for (int r = 0; r < 8; ++r) {
      const float send    = lo ? p1[r] : p0[r];
      const float partner = __shfl_xor(send, 16);
      Bp[r]     = (bf16_t)(lo ? p0[r] : partner);
      Bp[8 + r] = (bf16_t)(lo ? partner : p1[r]);
    }

    // ---- O^T += V^T * P (16 c-tiles, K = 32 j) ----
    #pragma unroll
    for (int ct = 0; ct < 16; ++ct) {
      const bf16_t* vr = &Vt[buf][ct * 16 + l16][0];
      const bf16x8 v0 = *(const bf16x8*)(vr + (lo ? 0 : 16));
      const bf16x8 v1 = *(const bf16x8*)(vr + 8 + (lo ? 0 : 16));
      bf16x16 Av;
      #pragma unroll
      for (int e = 0; e < 8; ++e) { Av[e] = v0[e]; Av[8 + e] = v1[e]; }
      Ov[ct] = __builtin_amdgcn_wmma_f32_16x16x32_bf16(false, Av, false, Bp,
                                                       (short)0, Ov[ct], false, false);
    }

#if USE_TDM
    if (wave == 0) __builtin_amdgcn_s_wait_tensorcnt(0);
    __syncthreads();   // next buffer visible to all waves
#endif
  }

  // ---- epilogue: out = gamma * (O / l) + x ----
  const float g    = gamma[0];
  const float linv = 1.0f / l;
  #pragma unroll
  for (int ct = 0; ct < 16; ++ct) {
    #pragma unroll
    for (int r = 0; r < 8; ++r) {
      const int    c   = ct * 16 + aoff + r;
      const size_t idx = ((size_t)b * C_ + c) * N_ + icol;
      out[idx] = g * (Ov[ct][r] * linv) + x[idx];
    }
  }
}

// ---------------------------------------------------------------------------
// Launch. Workspace: qT (2 MB) | kT (2 MB) | v bf16 (16 MB) = 20 MB.
// ---------------------------------------------------------------------------
extern "C" void kernel_launch(void* const* d_in, const int* in_sizes, int n_in,
                              void* d_out, int out_size, void* d_ws, size_t ws_size,
                              hipStream_t stream) {
  const float* x     = (const float*)d_in[0];
  const float* Wq    = (const float*)d_in[1];
  const float* bq    = (const float*)d_in[2];
  const float* Wk    = (const float*)d_in[3];
  const float* bk    = (const float*)d_in[4];
  const float* Wv    = (const float*)d_in[5];
  const float* bv    = (const float*)d_in[6];
  const float* gamma = (const float*)d_in[7];
  float* out = (float*)d_out;

  bf16_t* qT  = (bf16_t*)d_ws;
  bf16_t* kT  = qT + (size_t)B_ * N_ * C8_;
  bf16_t* vbf = kT + (size_t)B_ * N_ * C8_;

  dim3 grid(N_ / 64, B_);
  qkv_proj_kernel<<<grid, 128, 0, stream>>>(x, Wq, bq, Wk, bk, Wv, bv, qT, kT, vbf);
  attn_kernel<<<grid, 128, 0, stream>>>(qT, kT, vbf, x, gamma, out);
}
